// One_Shot_same_Loss_24773371363367
// MI455X (gfx1250) — matrix operand order repaired and checked by reading, not compiled, not run on hardware
//
#include <hip/hip_runtime.h>
#include <hip/hip_bf16.h>

// ---------------------------------------------------------------------------
// Pairwise-distance masked-mean loss for MI455X (gfx1250).
// Gram matrix via V_WMMA_F32_16X16X32_BF16 with fp32 = bf16(hi)+bf16(lo) split:
//   dot ~= hi*hi + hi*lo + lo*hi   (3 WMMAs per K=32 chunk, f32 accumulate)
// Each wave register-blocks a 32x32 output region (2x2 WMMA tiles): per K
// chunk, 8 operand loads feed 12 WMMAs (0.67 loads/WMMA vs 1.33 before).
// ---------------------------------------------------------------------------

#define NROWS 8192
#define DDIM  256
#define KCHUNKS (DDIM / 32)     // 8
#define NBI (NROWS / 64)        // 128 row blocks (64 rows each)
#define NBJ (NROWS / 128)       // 64  col blocks (128 cols each)

typedef __attribute__((ext_vector_type(16))) __bf16 v16bf;
typedef __attribute__((ext_vector_type(8)))  float  v8f;

__device__ __forceinline__ unsigned short f32_to_bf16_rne(float f) {
    unsigned int u = __float_as_uint(f);
    unsigned int r = u + 0x7FFFu + ((u >> 16) & 1u);   // round-to-nearest-even
    return (unsigned short)(r >> 16);
}
__device__ __forceinline__ float bf16_to_f32(unsigned short b) {
    return __uint_as_float(((unsigned int)b) << 16);
}

// ---------------------------------------------------------------------------
// Pre-swizzle fp32 P[8192][256] into hi/lo bf16 operand arrays laid out exactly
// as the WMMA A (16x32) and B (32x16) ISA VGPR layouts expect, so the hot
// kernel's per-lane operand load is a single contiguous 32-byte vector.
//
// Storage: [rowBlock(512)][kChunk(8)][lane(32)][elem(16)] ushort.
//
// A layout (16-bit 16x32): lane = m + 16*((kk>>3)&1); elem = (kk>>4)*8 + (kk&7)
// B layout (16-bit 32x16): lane = n + 16*((kk>>3)&1); elem = (kk>>4)*8 + (kk&7)
// ---------------------------------------------------------------------------
__global__ __launch_bounds__(DDIM)
void prep_split(const float* __restrict__ P,
                unsigned short* __restrict__ Ahi, unsigned short* __restrict__ Alo,
                unsigned short* __restrict__ Bhi, unsigned short* __restrict__ Blo) {
    const int i = blockIdx.x;      // row 0..8191
    const int k = threadIdx.x;     // col 0..255
    const float p = P[(size_t)i * DDIM + k];

    const unsigned short hb = f32_to_bf16_rne(p);
    const float hf = bf16_to_f32(hb);
    const unsigned short lb = f32_to_bf16_rne(p - hf);

    const int rb = i >> 4;         // 16-row block
    const int m  = i & 15;         // row within block (also n for B operand)
    const int c  = k >> 5;         // K chunk of 32
    const int kk = k & 31;

    const int laneA = m + (((kk >> 3) & 1) << 4);
    const int idxA  = ((kk >> 4) << 3) + (kk & 7);
    const size_t offA = (((size_t)(rb * KCHUNKS + c) * 32 + laneA) << 4) + idxA;

    const int laneB = m + (((kk >> 3) & 1) << 4);
    const int idxB  = ((kk >> 4) << 3) + (kk & 7);
    const size_t offB = (((size_t)(rb * KCHUNKS + c) * 32 + laneB) << 4) + idxB;

    Ahi[offA] = hb;  Alo[offA] = lb;
    Bhi[offB] = hb;  Blo[offB] = lb;
}

// ---------------------------------------------------------------------------
__global__ __launch_bounds__(256)
void sqnorm_kernel(const float* __restrict__ P, float* __restrict__ sq) {
    const int i = blockIdx.x * blockDim.x + threadIdx.x;
    const float4* row = (const float4*)(P + (size_t)i * DDIM);
    float s = 0.f;
#pragma unroll 8
    for (int k = 0; k < DDIM / 4; ++k) {
        float4 v = row[k];
        s += v.x * v.x + v.y * v.y + v.z * v.z + v.w * v.w;
    }
    sq[i] = s;
}

// ---------------------------------------------------------------------------
__global__ void zero_accum(float* s, unsigned int* c) { *s = 0.f; *c = 0u; }

__global__ void finalize(const float* s, const unsigned int* c, float* out) {
    out[0] = (s[0] / (float)c[0]) * 1.0f;   // WEIGHT = 1.0
}

// ---------------------------------------------------------------------------
// Workgroup (256 threads = 8 wave32, arranged 2x4) computes a 64x128 block of
// the distance matrix. Each wave owns a 32x32 register block = 2x2 WMMA tiles
// with four f32 accumulators. Fully sub-diagonal work is skipped at block and
// at wave granularity (wave-uniform branches keep EXEC all-ones for WMMA).
// ---------------------------------------------------------------------------
__global__ __launch_bounds__(256)
void pair_dist_kernel(const unsigned short* __restrict__ Ahi,
                      const unsigned short* __restrict__ Alo,
                      const unsigned short* __restrict__ Bhi,
                      const unsigned short* __restrict__ Blo,
                      const float* __restrict__ sq,
                      const int* __restrict__ targets,
                      float* __restrict__ sumAcc,
                      unsigned int* __restrict__ cntAcc) {
    const int TI = blockIdx.y;     // 64-row block (rows i)
    const int TJ = blockIdx.x;     // 128-col block (cols j)
    // Block rows start at TI*64, cols end at TJ*128+127. Skip fully-lower blocks.
    if (TI * 64 >= (TJ + 1) * 128) return;

    const int wave = threadIdx.x >> 5;
    const int lane = threadIdx.x & 31;
    const int rw = wave >> 2;            // 0..1: 32-row strip within block
    const int cw = wave & 3;             // 0..3: 32-col strip within block

    const int rowBase = TI * 64 + rw * 32;
    const int colBase = TJ * 128 + cw * 32;

    float        localSum = 0.f;
    unsigned int localCnt = 0u;

    // Wave-uniform: any gi < gj in this 32x32 region?
    if (rowBase < colBase + 32) {
        const int rbA0 = TI * 4 + rw * 2;        // 16-row block indices (A rows)
        const int rbB0 = TJ * 8 + cw * 2;        // 16-row block indices (B cols)

        v8f acc[2][2] = {};
#pragma unroll
        for (int c = 0; c < KCHUNKS; ++c) {
            const size_t oa0 = ((size_t)((rbA0 + 0) * KCHUNKS + c) * 32 + lane) << 4;
            const size_t oa1 = ((size_t)((rbA0 + 1) * KCHUNKS + c) * 32 + lane) << 4;
            const size_t ob0 = ((size_t)((rbB0 + 0) * KCHUNKS + c) * 32 + lane) << 4;
            const size_t ob1 = ((size_t)((rbB0 + 1) * KCHUNKS + c) * 32 + lane) << 4;
            const v16bf ah0 = *(const v16bf*)(Ahi + oa0);
            const v16bf al0 = *(const v16bf*)(Alo + oa0);
            const v16bf ah1 = *(const v16bf*)(Ahi + oa1);
            const v16bf al1 = *(const v16bf*)(Alo + oa1);
            const v16bf bh0 = *(const v16bf*)(Bhi + ob0);
            const v16bf bl0 = *(const v16bf*)(Blo + ob0);
            const v16bf bh1 = *(const v16bf*)(Bhi + ob1);
            const v16bf bl1 = *(const v16bf*)(Blo + ob1);

            acc[0][0] = __builtin_amdgcn_wmma_f32_16x16x32_bf16(
                            false, ah0, false, bh0, (short)0, acc[0][0], false, false);
            acc[0][0] = __builtin_amdgcn_wmma_f32_16x16x32_bf16(
                            false, ah0, false, bl0, (short)0, acc[0][0], false, false);
            acc[0][0] = __builtin_amdgcn_wmma_f32_16x16x32_bf16(
                            false, al0, false, bh0, (short)0, acc[0][0], false, false);

            acc[0][1] = __builtin_amdgcn_wmma_f32_16x16x32_bf16(
                            false, ah0, false, bh1, (short)0, acc[0][1], false, false);
            acc[0][1] = __builtin_amdgcn_wmma_f32_16x16x32_bf16(
                            false, ah0, false, bl1, (short)0, acc[0][1], false, false);
            acc[0][1] = __builtin_amdgcn_wmma_f32_16x16x32_bf16(
                            false, al0, false, bh1, (short)0, acc[0][1], false, false);

            acc[1][0] = __builtin_amdgcn_wmma_f32_16x16x32_bf16(
                            false, ah1, false, bh0, (short)0, acc[1][0], false, false);
            acc[1][0] = __builtin_amdgcn_wmma_f32_16x16x32_bf16(
                            false, ah1, false, bl0, (short)0, acc[1][0], false, false);
            acc[1][0] = __builtin_amdgcn_wmma_f32_16x16x32_bf16(
                            false, al1, false, bh0, (short)0, acc[1][0], false, false);

            acc[1][1] = __builtin_amdgcn_wmma_f32_16x16x32_bf16(
                            false, ah1, false, bh1, (short)0, acc[1][1], false, false);
            acc[1][1] = __builtin_amdgcn_wmma_f32_16x16x32_bf16(
                            false, ah1, false, bl1, (short)0, acc[1][1], false, false);
            acc[1][1] = __builtin_amdgcn_wmma_f32_16x16x32_bf16(
                            false, al1, false, bh1, (short)0, acc[1][1], false, false);
        }

        // Epilogue. C layout: VGPR r, lane l -> n = l&15, m = r + 8*(l>>4).
        const int nn  = lane & 15;
        const int hi8 = (lane >> 4) << 3;
#pragma unroll
        for (int a = 0; a < 2; ++a) {
            const int miBase = rowBase + a * 16 + hi8;
            float sqi[8];
            int   ti[8];
#pragma unroll
            for (int r = 0; r < 8; ++r) {
                sqi[r] = sq[miBase + r];
                ti[r]  = targets[miBase + r];
            }
#pragma unroll
            for (int b = 0; b < 2; ++b) {
                const int gj = colBase + b * 16 + nn;
                const int tj = targets[gj];
                const float sqj = sq[gj];
#pragma unroll
                for (int r = 0; r < 8; ++r) {
                    const int gi = miBase + r;
                    if (gi < gj && ti[r] == tj) {
                        const float d2 = sqi[r] + sqj - 2.0f * acc[a][b][r];
                        localSum += sqrtf(fmaxf(d2, 0.f));
                        localCnt += 1u;
                    }
                }
            }
        }
    }

    // wave32 reduction, then one atomic per wave
#pragma unroll
    for (int off = 16; off > 0; off >>= 1) {
        localSum += __shfl_down(localSum, off, 32);
        localCnt += __shfl_down(localCnt, off, 32);
    }
    if (lane == 0) {
        atomicAdd(sumAcc, localSum);
        atomicAdd(cntAcc, localCnt);
    }
}

// ---------------------------------------------------------------------------
extern "C" void kernel_launch(void* const* d_in, const int* in_sizes, int n_in,
                              void* d_out, int out_size, void* d_ws, size_t ws_size,
                              hipStream_t stream) {
    const float* P       = (const float*)d_in[0];
    const int*   targets = (const int*)d_in[1];

    char* ws = (char*)d_ws;
    const size_t MAT = (size_t)NROWS * DDIM * sizeof(unsigned short);  // 4 MB

    float*          sumAcc = (float*)ws;
    unsigned int*   cntAcc = (unsigned int*)(ws + 8);
    unsigned short* Ahi    = (unsigned short*)(ws + 64);
    unsigned short* Alo    = (unsigned short*)(ws + 64 + 1 * MAT);
    unsigned short* Bhi    = (unsigned short*)(ws + 64 + 2 * MAT);
    unsigned short* Blo    = (unsigned short*)(ws + 64 + 3 * MAT);
    float*          sq     = (float*)(ws + 64 + 4 * MAT);

    hipLaunchKernelGGL(zero_accum, dim3(1), dim3(1), 0, stream, sumAcc, cntAcc);
    hipLaunchKernelGGL(prep_split, dim3(NROWS), dim3(DDIM), 0, stream,
                       P, Ahi, Alo, Bhi, Blo);
    hipLaunchKernelGGL(sqnorm_kernel, dim3(NROWS / 256), dim3(256), 0, stream, P, sq);
    hipLaunchKernelGGL(pair_dist_kernel, dim3(NBJ, NBI), dim3(256), 0, stream,
                       Ahi, Alo, Bhi, Blo, sq, targets, sumAcc, cntAcc);
    hipLaunchKernelGGL(finalize, dim3(1), dim3(1), 0, stream,
                       sumAcc, cntAcc, (float*)d_out);
}
